// SINQSparseLinear_64330020159913
// MI455X (gfx1250) — matrix operand order repaired and checked by reading, not compiled, not run on hardware
//
#include <hip/hip_runtime.h>

#define TOK     8192
#define IN_DIM  4096
#define OUT_DIM 4096

#define BM 128            // block tile M (2 waves)
#define BN 256            // block tile N (4 waves)
#define BK 32             // K step per stage
#define NKT (IN_DIM / BK) // 128 K-steps
#define LDS_STRIDE 40     // 32 cols + 8 pad bf16 -> 80 B row stride (bank-friendly)

typedef __attribute__((ext_vector_type(16))) __bf16       bf16x16;
typedef __attribute__((ext_vector_type(8)))  __bf16       bf16x8;
typedef __attribute__((ext_vector_type(8)))  float        f32x8;
typedef __attribute__((ext_vector_type(4)))  float        f32x4;
typedef __attribute__((ext_vector_type(4)))  int          i32x4;
typedef __attribute__((ext_vector_type(4)))  unsigned int u32x4;

union Frag32B {
    u32x4   q[2];
    bf16x16 v;
};

// 16-byte async copy global -> LDS (tracked with ASYNCcnt, no VGPR round-trip)
__device__ __forceinline__ void async_copy_b128(const __bf16* gsrc, const __bf16* lds_dst) {
    unsigned l = (unsigned)(size_t)lds_dst;  // low 32 bits = LDS offset
    asm volatile("global_load_async_to_lds_b128 %0, %1, off"
                 :: "v"(l), "v"(gsrc) : "memory");
}
__device__ __forceinline__ void wait_async_all() {
    asm volatile("s_wait_asynccnt 0x0" ::: "memory");
}

// ---------------------------------------------------------------------------
// Pre-pass 1: dequantize W (int4 codes in int32) -> bf16, fused mask/scales.
// ---------------------------------------------------------------------------
__global__ __launch_bounds__(256) void dequant_w_kernel(
    const int* __restrict__ Wq, const float* __restrict__ scales,
    const float* __restrict__ zeros, const unsigned char* __restrict__ mask,
    const float* __restrict__ scale2, __bf16* __restrict__ Wb) {
    const int idx = (blockIdx.x * 256 + threadIdx.x) * 8;
    const int row = idx >> 12;          // / IN_DIM
    const int col = idx & (IN_DIM - 1);

    const float s = scales[row];
    const float z = zeros[row];

    i32x4 q0 = *(const i32x4*)(Wq + idx);
    i32x4 q1 = *(const i32x4*)(Wq + idx + 4);
    f32x4 s2a = *(const f32x4*)(scale2 + col);
    f32x4 s2b = *(const f32x4*)(scale2 + col + 4);
    unsigned long long mbits = *(const unsigned long long*)(mask + idx);

    bf16x8 o;
#pragma unroll
    for (int j = 0; j < 8; ++j) {
        float q  = (float)(j < 4 ? q0[j] : q1[j - 4]);
        float s2 = (j < 4 ? s2a[j] : s2b[j - 4]);
        float m  = (float)((mbits >> (8 * j)) & 1ull);
        o[j] = (__bf16)((q - z) * s * s2 * m);
    }
    *(bf16x8*)(Wb + idx) = o;
}

// ---------------------------------------------------------------------------
// Pre-pass 2: x f32 -> bf16.
// ---------------------------------------------------------------------------
__global__ __launch_bounds__(256) void cvt_x_kernel(
    const float* __restrict__ X, __bf16* __restrict__ Xb) {
    const int idx = (blockIdx.x * 256 + threadIdx.x) * 8;
    f32x4 a = *(const f32x4*)(X + idx);
    f32x4 b = *(const f32x4*)(X + idx + 4);
    bf16x8 o;
#pragma unroll
    for (int j = 0; j < 8; ++j)
        o[j] = (__bf16)(j < 4 ? a[j] : b[j - 4]);
    *(bf16x8*)(Xb + idx) = o;
}

// ---------------------------------------------------------------------------
// WMMA bf16 GEMM with async-to-LDS double-buffered staging.
// Workgroup: 8 waves, 2(M) x 4(N); wave tile 64x64 = 4x4 WMMA accumulators.
// Per K=32 step: stage 128x32 X-tile (8 KB) + 256x32 W-tile (16 KB) into LDS
// via global_load_async_to_lds_b128; all 16 fragment ds_load_b128 issue up
// front so s_wait_dscnt staggers across the 16-WMMA burst.
// ---------------------------------------------------------------------------
__global__ __launch_bounds__(256) void wmma_gemm_kernel(
    const __bf16* __restrict__ Xb, const __bf16* __restrict__ Wb,
    const float* __restrict__ bias, float* __restrict__ out) {
    __shared__ __bf16 Xs[2][BM * LDS_STRIDE];   // 2 x 10240 B
    __shared__ __bf16 Ws[2][BN * LDS_STRIDE];   // 2 x 20480 B

    const int tid   = threadIdx.x;
    const int lane  = tid & 31;
    const int wave  = tid >> 5;
    const int waveM = wave >> 2;   // 0..1
    const int waveN = wave & 3;    // 0..3
    const int ln    = lane & 15;
    const int hi    = lane >> 4;   // 0 or 1

    const int blockM = blockIdx.y * BM;
    const int blockN = blockIdx.x * BN;

    // Staging: chunk task t -> row = t>>2, chunk = t&3 (4 x 16 B chunks/row).
    auto stage = [&](int kt, int buf) {
        const int k = kt * BK;
#pragma unroll
        for (int i = 0; i < 2; ++i) {
            const int t = tid + i * 256;
            const int row = t >> 2, chunk = t & 3;
            async_copy_b128(Xb + (size_t)(blockM + row) * IN_DIM + k + chunk * 8,
                            &Xs[buf][row * LDS_STRIDE + chunk * 8]);
        }
#pragma unroll
        for (int i = 0; i < 4; ++i) {
            const int t = tid + i * 256;
            const int row = t >> 2, chunk = t & 3;
            async_copy_b128(Wb + (size_t)(blockN + row) * IN_DIM + k + chunk * 8,
                            &Ws[buf][row * LDS_STRIDE + chunk * 8]);
        }
    };

    f32x8 acc[4][4] = {};

    stage(0, 0);
    wait_async_all();
    __syncthreads();

    for (int kt = 0; kt < NKT; ++kt) {
        const int buf = kt & 1;
        if (kt + 1 < NKT)
            stage(kt + 1, buf ^ 1);

        // Preload ALL fragments, then burst 16 WMMAs (waits overlap compute).
        Frag32B a[4], b[4];
#pragma unroll
        for (int m = 0; m < 4; ++m) {
            // A (16x32): lane row M=l%16; lanes<16 K{0..7,16..23}, else K{8..15,24..31}
            const __bf16* p = &Xs[buf][(waveM * 64 + m * 16 + ln) * LDS_STRIDE];
            a[m].q[0] = *(const u32x4*)(p + hi * 8);
            a[m].q[1] = *(const u32x4*)(p + 16 + hi * 8);
        }
#pragma unroll
        for (int n = 0; n < 4; ++n) {
            // B (32x16): lane col N=l%16 (a W row); lanes<16 K{0..7,16..23}, else K{8..15,24..31}
            const __bf16* p = &Ws[buf][(waveN * 64 + n * 16 + ln) * LDS_STRIDE + hi * 8];
            b[n].q[0] = *(const u32x4*)(p);
            b[n].q[1] = *(const u32x4*)(p + 16);
        }
#pragma unroll
        for (int n = 0; n < 4; ++n)
#pragma unroll
            for (int m = 0; m < 4; ++m)
                acc[m][n] = __builtin_amdgcn_wmma_f32_16x16x32_bf16(
                    /*neg_a=*/false, a[m].v, /*neg_b=*/false, b[n].v,
                    /*c_mod=*/(short)0, acc[m][n],
                    /*reuse_a=*/false, /*reuse_b=*/false);

        wait_async_all();     // staged writes for buf^1 landed (this wave)
        __syncthreads();      // all waves done reading buf & staging visible
    }

    // Epilogue: D tile (16x16 f32): VGPR r, lane l -> row r + 8*(l>=16), col l%16.
#pragma unroll
    for (int n = 0; n < 4; ++n) {
        const int col = blockN + waveN * 64 + n * 16 + ln;
        const float bv = bias[col];
#pragma unroll
        for (int m = 0; m < 4; ++m) {
            const int row0 = blockM + waveM * 64 + m * 16 + hi * 8;
#pragma unroll
            for (int r = 0; r < 8; ++r)
                out[(size_t)(row0 + r) * OUT_DIM + col] = acc[m][n][r] + bv;
        }
    }
}

// ---------------------------------------------------------------------------
extern "C" void kernel_launch(void* const* d_in, const int* in_sizes, int n_in,
                              void* d_out, int out_size, void* d_ws, size_t ws_size,
                              hipStream_t stream) {
    const float*         x      = (const float*)d_in[0];
    const int*           Wq     = (const int*)d_in[1];
    const float*         scales = (const float*)d_in[2];
    const float*         zeros  = (const float*)d_in[3];
    const unsigned char* mask   = (const unsigned char*)d_in[4];  // jnp.bool_ = 1 byte
    const float*         scale2 = (const float*)d_in[5];
    const float*         bias   = (const float*)d_in[6];
    float*               out    = (float*)d_out;

    __bf16* Wb = (__bf16*)d_ws;                        // 32 MiB
    __bf16* Xb = Wb + (size_t)OUT_DIM * IN_DIM;        // 64 MiB

    dequant_w_kernel<<<(OUT_DIM * IN_DIM) / (8 * 256), 256, 0, stream>>>(
        Wq, scales, zeros, mask, scale2, Wb);
    cvt_x_kernel<<<(TOK * IN_DIM) / (8 * 256), 256, 0, stream>>>(x, Xb);

    dim3 grid(OUT_DIM / BN, TOK / BM);  // (16, 64)
    wmma_gemm_kernel<<<grid, 256, 0, stream>>>(Xb, Wb, bias, out);
}
